// Glom_60413009985929
// MI455X (gfx1250) — compile-verified, matrix operand order checked by reading, not run on hardware
//
#include <hip/hip_runtime.h>

// ---------------------------------------------------------------------------
// GLOM on MI455X (gfx1250): bf16 WMMA feed-forwards + fused consensus.
// ---------------------------------------------------------------------------

typedef __bf16   v16bf __attribute__((ext_vector_type(16)));
typedef float    v8f   __attribute__((ext_vector_type(8)));
typedef unsigned int v4u __attribute__((ext_vector_type(4)));

union FragBF16 { v16bf v; v4u q[2]; };

__device__ __forceinline__ unsigned short f2bf(float f) {
    unsigned int u = __float_as_uint(f);
    unsigned int r = u + 0x7FFFu + ((u >> 16) & 1u);   // round-to-nearest-even
    return (unsigned short)(r >> 16);
}

// Load a 16x32 bf16 fragment (A-layout per CDNA5 ISA 7.12.2):
// lane&15 = row, lane>>4 selects K-halves {0..7,16..23} vs {8..15,24..31}.
__device__ __forceinline__ v16bf load_frag(const unsigned short* __restrict__ base,
                                           int ld, int r0, int k0, int lane) {
    int r  = lane & 15;
    int hi = (lane >> 4) & 1;
    const unsigned short* p = base + (size_t)(r0 + r) * ld + k0 + hi * 8;
    FragBF16 f;
    f.q[0] = *(const v4u*)(p);        // K = k0 + hi*8 .. +7        (16B load)
    f.q[1] = *(const v4u*)(p + 16);   // K = k0 + 16 + hi*8 .. +7   (16B load)
    return f.v;
}

// ---------------------------------------------------------------------------
// Small helper kernels
// ---------------------------------------------------------------------------

__global__ void convert_bf16_kernel(const float* __restrict__ src,
                                    unsigned short* __restrict__ dst, int count) {
    int idx = blockIdx.x * 256 + threadIdx.x;
    if (idx < count) dst[idx] = f2bf(src[idx]);
}

// tokens[b,n,o] = patch(img) @ W + bias   (K=196, small -> scalar FMA)
__global__ void tokens_kernel(const float* __restrict__ img,
                              const float* __restrict__ w,
                              const float* __restrict__ bias,
                              float* __restrict__ tokens) {
    size_t idx = (size_t)blockIdx.x * 256 + threadIdx.x;  // 16*256*512
    int o  = (int)(idx & 511);
    int tn = (int)((idx >> 9) & 255);
    int b  = (int)(idx >> 17);
    int ph = tn >> 4, pw = tn & 15;
    const float* ip = img + ((size_t)b * 224 + (size_t)ph * 14) * 224 + pw * 14;
    float s = bias[o];
    #pragma unroll 2
    for (int a = 0; a < 14; ++a)
        for (int c = 0; c < 14; ++c)
            s = fmaf(ip[a * 224 + c], w[(a * 14 + c) * 512 + o], s);
    tokens[idx] = s;
}

// levels[b,n,l,d] = init_levels[l,d]
__global__ void init_levels_kernel(const float* __restrict__ init_levels,
                                   float* __restrict__ levels) {
    size_t idx = (size_t)blockIdx.x * 256 + threadIdx.x;  // 16*256*6*512
    levels[idx] = init_levels[idx % 3072];
}

// X_bu level 0 = bf16(tokens)   (constant across iterations)
__global__ void xbu0_kernel(const float* __restrict__ tokens,
                            unsigned short* __restrict__ xbu) {
    size_t idx = (size_t)blockIdx.x * 256 + threadIdx.x;  // 4096*512
    int d = (int)(idx & 511);
    size_t i = idx >> 9;
    xbu[i * 3072 + d] = f2bf(tokens[idx]);
}

// X_bu levels 1..5 = bf16(levels[0..4])
__global__ void prep_bu_kernel(const float* __restrict__ levels,
                               unsigned short* __restrict__ xbu) {
    size_t idx = (size_t)blockIdx.x * 256 + threadIdx.x;  // 4096*5*512
    int d = (int)(idx & 511);
    int j = (int)((idx >> 9) % 5);        // 0..4 -> dest level j+1
    size_t i = idx / 2560;
    xbu[i * 3072 + (size_t)(j + 1) * 512 + d] =
        f2bf(levels[i * 3072 + (size_t)j * 512 + d]);
}

// X_td level j = bf16(levels[j+1] + pos)
__global__ void prep_td_kernel(const float* __restrict__ levels,
                               const float* __restrict__ pos,
                               unsigned short* __restrict__ xtd) {
    size_t idx = (size_t)blockIdx.x * 256 + threadIdx.x;  // 4096*5*512
    int d = (int)(idx & 511);
    int j = (int)((idx >> 9) % 5);
    size_t i = idx / 2560;
    int n = (int)(i & 255);
    xtd[i * 2560 + (size_t)j * 512 + d] =
        f2bf(levels[i * 3072 + (size_t)(j + 1) * 512 + d] + pos[(size_t)n * 512 + d]);
}

// rnorm[row] = 1 / max(||levels_row||, 1e-12); one wave (32 lanes) per row
__global__ void norms_kernel(const float* __restrict__ levels,
                             float* __restrict__ rnorm) {
    int wave = threadIdx.x >> 5, lane = threadIdx.x & 31;
    size_t row = (size_t)blockIdx.x * 8 + wave;           // 24576 rows
    const float* p = levels + row * 512;
    float s = 0.0f;
    for (int d = lane; d < 512; d += 32) { float v = p[d]; s = fmaf(v, v, s); }
    for (int m = 16; m; m >>= 1) s += __shfl_xor(s, m, 32);
    if (lane == 0) rnorm[row] = 1.0f / fmaxf(sqrtf(s), 1e-12f);
}

// levels = acc / contrib
__global__ void finalize_kernel(const float* __restrict__ acc,
                                float* __restrict__ levels) {
    size_t idx = (size_t)blockIdx.x * 256 + threadIdx.x;
    int l = (int)((idx >> 9) % 6);
    levels[idx] = acc[idx] * (l == 5 ? (1.0f / 3.0f) : 0.25f);
}

// ---------------------------------------------------------------------------
// Fused consensus: per (b, l, 16-query block). Writes acc = levels + cons.
// ---------------------------------------------------------------------------
__global__ __launch_bounds__(256) void consensus_kernel(
        const float* __restrict__ levels, const float* __restrict__ rnorm,
        float* __restrict__ acc) {
    __shared__ float qs[16][512];
    __shared__ float sm[16][257];
    const int t = threadIdx.x;
    const int ib = blockIdx.x, l = blockIdx.y, b = blockIdx.z;
    const int i0 = ib * 16;

    for (int idx = t; idx < 16 * 512; idx += 256) {
        int k = idx >> 9, d = idx & 511;
        qs[k][d] = levels[(((size_t)(b * 256 + i0 + k)) * 6 + l) * 512 + d];
    }
    __syncthreads();

    {   // sim row for key j = t
        const int j = t;
        float simk[16];
        #pragma unroll
        for (int k = 0; k < 16; ++k) simk[k] = 0.0f;
        const float* krow = levels + (((size_t)(b * 256 + j)) * 6 + l) * 512;
        for (int d = 0; d < 512; ++d) {
            float kv = krow[d];
            #pragma unroll
            for (int k = 0; k < 16; ++k) simk[k] = fmaf(qs[k][d], kv, simk[k]);
        }
        float rn = rnorm[((size_t)(b * 256 + j)) * 6 + l] * 0.04419417382415922f;
        #pragma unroll
        for (int k = 0; k < 16; ++k) {
            float s = simk[k] * rn;
            if (j == i0 + k) s = -0.0005f;   // TOKEN_ATTEND_SELF_VALUE
            sm[k][j] = s;
        }
    }
    __syncthreads();

    if (t < 16) {   // softmax over j for query t
        const int k = t;
        float m = -1e30f;
        for (int j = 0; j < 256; ++j) m = fmaxf(m, sm[k][j]);
        float sum = 0.0f;
        for (int j = 0; j < 256; ++j) { float e = __expf(sm[k][j] - m); sm[k][j] = e; sum += e; }
        float inv = 1.0f / sum;
        for (int j = 0; j < 256; ++j) sm[k][j] *= inv;
    }
    __syncthreads();

    {   // apply: this thread owns dims t and t+256
        float o0[16], o1[16];
        #pragma unroll
        for (int k = 0; k < 16; ++k) { o0[k] = 0.0f; o1[k] = 0.0f; }
        for (int j = 0; j < 256; ++j) {
            const float* krow = levels + (((size_t)(b * 256 + j)) * 6 + l) * 512;
            float v0 = krow[t], v1 = krow[t + 256];
            #pragma unroll
            for (int k = 0; k < 16; ++k) {
                o0[k] = fmaf(sm[k][j], v0, o0[k]);
                o1[k] = fmaf(sm[k][j], v1, o1[k]);
            }
        }
        #pragma unroll
        for (int k = 0; k < 16; ++k) {
            size_t base = (((size_t)(b * 256 + i0 + k)) * 6 + l) * 512;
            acc[base + t]       = levels[base + t]       + o0[k];
            acc[base + t + 256] = levels[base + t + 256] + o1[k];
        }
    }
}

// ---------------------------------------------------------------------------
// WMMA GEMMs. 256 threads = 8 waves; wave tile 32x64 (2x4 wmma frags);
// block tile 64x256; K-step 32 with v_wmma_f32_16x16x32_bf16.
// ---------------------------------------------------------------------------

// H[4096,2048] = gelu(A[4096,K] @ W[2048,K]^T + b1), stored bf16
__global__ __launch_bounds__(256) void ff_gemm1(
        const unsigned short* __restrict__ A, int lda,
        const unsigned short* __restrict__ W,
        const float* __restrict__ bias,
        unsigned short* __restrict__ H, int K) {
    const int lane = threadIdx.x & 31, wave = threadIdx.x >> 5;
    const int row0 = blockIdx.x * 64 + (wave >> 2) * 32;
    const int col0 = blockIdx.y * 256 + (wave & 3) * 64;

    v8f c[2][4];
    #pragma unroll
    for (int mi = 0; mi < 2; ++mi)
        #pragma unroll
        for (int ni = 0; ni < 4; ++ni)
            #pragma unroll
            for (int e = 0; e < 8; ++e) c[mi][ni][e] = 0.0f;

    for (int k0 = 0; k0 < K; k0 += 32) {
        v16bf a[2], bfr[4];
        #pragma unroll
        for (int mi = 0; mi < 2; ++mi) a[mi]  = load_frag(A, lda, row0 + mi * 16, k0, lane);
        #pragma unroll
        for (int ni = 0; ni < 4; ++ni) bfr[ni] = load_frag(W, K, col0 + ni * 16, k0, lane);
        #pragma unroll
        for (int mi = 0; mi < 2; ++mi)
            #pragma unroll
            for (int ni = 0; ni < 4; ++ni)
                c[mi][ni] = __builtin_amdgcn_wmma_f32_16x16x32_bf16(
                    false, a[mi], false, bfr[ni], (short)0, c[mi][ni], false, false);
    }

    const int cn = lane & 15, rh = (lane >> 4) * 8;
    #pragma unroll
    for (int mi = 0; mi < 2; ++mi)
        #pragma unroll
        for (int ni = 0; ni < 4; ++ni) {
            int col = col0 + ni * 16 + cn;
            float bv = bias[col];
            #pragma unroll
            for (int e = 0; e < 8; ++e) {
                int row = row0 + mi * 16 + rh + e;
                float x = c[mi][ni][e] + bv;
                float g = 0.5f * x * (1.0f + erff(x * 0.70710678118654752f));  // exact gelu
                H[(size_t)row * 2048 + col] = f2bf(g);
            }
        }
}

// acc[row, col] += A[4096,K] @ W[512,K]^T + b2   (acc pre-offset by level)
__global__ __launch_bounds__(256) void ff_gemm2(
        const unsigned short* __restrict__ A, int lda,
        const unsigned short* __restrict__ W,
        const float* __restrict__ bias,
        float* __restrict__ accBase, int accStride, int K) {
    const int lane = threadIdx.x & 31, wave = threadIdx.x >> 5;
    const int row0 = blockIdx.x * 64 + (wave >> 2) * 32;
    const int col0 = blockIdx.y * 256 + (wave & 3) * 64;

    v8f c[2][4];
    #pragma unroll
    for (int mi = 0; mi < 2; ++mi)
        #pragma unroll
        for (int ni = 0; ni < 4; ++ni)
            #pragma unroll
            for (int e = 0; e < 8; ++e) c[mi][ni][e] = 0.0f;

    for (int k0 = 0; k0 < K; k0 += 32) {
        v16bf a[2], bfr[4];
        #pragma unroll
        for (int mi = 0; mi < 2; ++mi) a[mi]  = load_frag(A, lda, row0 + mi * 16, k0, lane);
        #pragma unroll
        for (int ni = 0; ni < 4; ++ni) bfr[ni] = load_frag(W, K, col0 + ni * 16, k0, lane);
        #pragma unroll
        for (int mi = 0; mi < 2; ++mi)
            #pragma unroll
            for (int ni = 0; ni < 4; ++ni)
                c[mi][ni] = __builtin_amdgcn_wmma_f32_16x16x32_bf16(
                    false, a[mi], false, bfr[ni], (short)0, c[mi][ni], false, false);
    }

    const int cn = lane & 15, rh = (lane >> 4) * 8;
    #pragma unroll
    for (int mi = 0; mi < 2; ++mi)
        #pragma unroll
        for (int ni = 0; ni < 4; ++ni) {
            int col = col0 + ni * 16 + cn;
            float bv = bias[col];
            #pragma unroll
            for (int e = 0; e < 8; ++e) {
                int row = row0 + mi * 16 + rh + e;
                accBase[(size_t)row * accStride + col] += c[mi][ni][e] + bv;
            }
        }
}

// ---------------------------------------------------------------------------
// Host orchestration
// ---------------------------------------------------------------------------
extern "C" void kernel_launch(void* const* d_in, const int* in_sizes, int n_in,
                              void* d_out, int out_size, void* d_ws, size_t ws_size,
                              hipStream_t stream) {
    const float* img        = (const float*)d_in[0];
    const float* to_tok_w   = (const float*)d_in[1];
    const float* to_tok_b   = (const float*)d_in[2];
    const float* pos_emb    = (const float*)d_in[3];
    const float* init_lv    = (const float*)d_in[4];
    const float* bu_w1      = (const float*)d_in[5];
    const float* bu_b1      = (const float*)d_in[6];
    const float* bu_w2      = (const float*)d_in[7];
    const float* bu_b2      = (const float*)d_in[8];
    const float* td_w1      = (const float*)d_in[9];
    const float* td_b1      = (const float*)d_in[10];
    const float* td_w2      = (const float*)d_in[11];
    const float* td_b2      = (const float*)d_in[12];
    float* levels = (float*)d_out;   // (16,256,6,512) fp32 — lives in d_out

    // workspace carve-up (256B aligned)
    char* ws = (char*)d_ws;
    size_t off = 0;
    auto carve = [&](size_t bytes) -> char* {
        off = (off + 255) & ~(size_t)255;
        char* p = ws + off; off += bytes; return p;
    };
    const size_t W1SZ = (size_t)6 * 2048 * 512;   // per-set element counts
    const size_t T1SZ = (size_t)5 * 2048 * 512;
    unsigned short* wbu1 = (unsigned short*)carve(W1SZ * 2);
    unsigned short* wbu2 = (unsigned short*)carve(W1SZ * 2);
    unsigned short* wtd1 = (unsigned short*)carve(T1SZ * 2);
    unsigned short* wtd2 = (unsigned short*)carve(T1SZ * 2);
    float*          tokens = (float*)carve((size_t)4096 * 512 * 4);
    unsigned short* xbu  = (unsigned short*)carve((size_t)4096 * 6 * 512 * 2);
    unsigned short* xtd  = (unsigned short*)carve((size_t)4096 * 5 * 512 * 2);
    unsigned short* Hbuf = (unsigned short*)carve((size_t)4096 * 2048 * 2);
    float*          acc  = (float*)carve((size_t)4096 * 6 * 512 * 4);
    float*          rnorm= (float*)carve((size_t)24576 * 4);
    (void)ws_size; (void)in_sizes; (void)n_in; (void)out_size;

    // 1) weights -> bf16 (once per call)
    convert_bf16_kernel<<<(int)(W1SZ / 256), 256, 0, stream>>>(bu_w1, wbu1, (int)W1SZ);
    convert_bf16_kernel<<<(int)(W1SZ / 256), 256, 0, stream>>>(bu_w2, wbu2, (int)W1SZ);
    convert_bf16_kernel<<<(int)(T1SZ / 256), 256, 0, stream>>>(td_w1, wtd1, (int)T1SZ);
    convert_bf16_kernel<<<(int)(T1SZ / 256), 256, 0, stream>>>(td_w2, wtd2, (int)T1SZ);

    // 2) patchify + token projection; init levels; bottom slice of X_bu
    tokens_kernel<<<8192, 256, 0, stream>>>(img, to_tok_w, to_tok_b, tokens);
    init_levels_kernel<<<49152, 256, 0, stream>>>(init_lv, levels);
    xbu0_kernel<<<8192, 256, 0, stream>>>(tokens, xbu);

    // 3) 12 GLOM iterations
    for (int it = 0; it < 12; ++it) {
        prep_bu_kernel<<<40960, 256, 0, stream>>>(levels, xbu);
        prep_td_kernel<<<40960, 256, 0, stream>>>(levels, pos_emb, xtd);
        norms_kernel<<<3072, 256, 0, stream>>>(levels, rnorm);
        // acc = levels + consensus(levels)
        consensus_kernel<<<dim3(16, 6, 16), 256, 0, stream>>>(levels, rnorm, acc);

        // bottom-up FF: level l input = xbu[:, l, :], output adds to acc level l
        for (int l = 0; l < 6; ++l) {
            ff_gemm1<<<dim3(64, 8), 256, 0, stream>>>(
                xbu + (size_t)l * 512, 3072,
                wbu1 + (size_t)l * 2048 * 512, bu_b1 + (size_t)l * 2048,
                Hbuf, 512);
            ff_gemm2<<<dim3(64, 2), 256, 0, stream>>>(
                Hbuf, 2048,
                wbu2 + (size_t)l * 512 * 2048, bu_b2 + (size_t)l * 512,
                acc + (size_t)l * 512, 3072, 2048);
        }
        // top-down FF: input level j = levels[j+1]+pos, output adds to acc level j
        for (int j = 0; j < 5; ++j) {
            ff_gemm1<<<dim3(64, 8), 256, 0, stream>>>(
                xtd + (size_t)j * 512, 2560,
                wtd1 + (size_t)j * 2048 * 512, td_b1 + (size_t)j * 2048,
                Hbuf, 512);
            ff_gemm2<<<dim3(64, 2), 256, 0, stream>>>(
                Hbuf, 2048,
                wtd2 + (size_t)j * 512 * 2048, td_b2 + (size_t)j * 512,
                acc + (size_t)j * 512, 3072, 2048);
        }
        finalize_kernel<<<49152, 256, 0, stream>>>(acc, levels);
    }
}
